// GNNEncoder_16355235463217
// MI455X (gfx1250) — matrix operand order repaired and checked by reading, not compiled, not run on hardware
//
#include <hip/hip_runtime.h>

// ---------------------------------------------------------------------------
// Fused GNN encoder for MI455X (gfx1250, wave32, WMMA 16x16x32 f16 -> f32 acc)
// ---------------------------------------------------------------------------

typedef __attribute__((ext_vector_type(16))) _Float16 v16h;
typedef __attribute__((ext_vector_type(8)))  _Float16 v8h;
typedef __attribute__((ext_vector_type(4)))  _Float16 v4h;
typedef __attribute__((ext_vector_type(8)))  float    v8f;

#define B_  4
#define N_  20000
#define E_  640000
#define ND  64     // NODE_DIM
#define ED  32     // EDGE_DIM
#define HD  64     // HIDDEN_DIM
#define F1  160    // 2*ND + ED  (msg MLP input)
#define F2  128    // ND + HD    (update MLP input)

union V16HU { v16h v; v8h h[2]; };

// A-fragment (16-bit A 16x32 layout): lane covers row M = lane&15.
// elems 0..7  -> K = kBase + halfSel*8 + e
// elems 8..15 -> K = kBase + 16 + halfSel*8 + (e-8)
static __device__ inline v16h ldsA(const _Float16* row, int kBase, int hs) {
  V16HU r;
  r.h[0] = *(const v8h*)(row + kBase + hs * 8);
  r.h[1] = *(const v8h*)(row + kBase + 16 + hs * 8);
  return r.v;
}

// B-fragment (16-bit B 32x16 layout): lane covers col N = lane&15.
// elems 0..7  -> K = kBase + halfSel*8 + e
// elems 8..15 -> K = kBase + 16 + halfSel*8 + (e-8)
static __device__ inline v16h ldsB(const _Float16* row, int kBase, int hs) {
  V16HU r;
  r.h[0] = *(const v8h*)(row + kBase + hs * 8);
  r.h[1] = *(const v8h*)(row + kBase + 16 + hs * 8);
  return r.v;
}

static __device__ inline void cvt4(_Float16* dst, float4 s) {
  v4h t = { (_Float16)s.x, (_Float16)s.y, (_Float16)s.z, (_Float16)s.w };
  *(v4h*)dst = t;
}

// ---------------------------------------------------------------------------
// Edge kernel: per 16-edge tile per wave:
//   gather [node[recv] | node[sent] | edge_attr] -> f16 LDS tile (16x160)
//   h   = relu(X @ W1^T + b1)      (5 K-chunks x 4 N-chunks WMMA)
//   msg = h @ W2^T + b2            (2 K-chunks x 4 N-chunks WMMA)
//   atomic scatter-add msg rows into agg[b][recv]
// ---------------------------------------------------------------------------
__launch_bounds__(128)
__global__ void gnn_edge_kernel(const float* __restrict__ node,
                                const int*   __restrict__ eidx,
                                const float* __restrict__ eattr,
                                const float* __restrict__ W1,
                                const float* __restrict__ b1,
                                const float* __restrict__ W2,
                                const float* __restrict__ b2,
                                float* __restrict__ agg)
{
  __shared__ __align__(16) _Float16 sW1[HD * F1];     // 20 KB, [n][k] row-major
  __shared__ __align__(16) _Float16 sW2[HD * HD];     //  8 KB
  __shared__ __align__(16) _Float16 sX[4][16 * F1];   // 20 KB, per-wave input tile
  __shared__ __align__(16) _Float16 sH[4][16 * HD];   //  8 KB, per-wave hidden tile
  __shared__ int sR[4][16];

  const int tid = threadIdx.x;
  for (int i = tid; i < HD * F1; i += 128) sW1[i] = (_Float16)W1[i];
  for (int i = tid; i < HD * HD; i += 128) sW2[i] = (_Float16)W2[i];
  __syncthreads();

  const int lane = tid & 31;
  const int w    = tid >> 5;
  const int m    = lane & 15;   // row (M) / col (N) index within tile
  const int hs   = lane >> 4;   // lane half-select

  float bias1[4], bias2[4];
  for (int nc = 0; nc < 4; ++nc) {
    bias1[nc] = b1[nc * 16 + m];
    bias2[nc] = b2[nc * 16 + m];
  }

  const int tilesPerBatch = E_ / 16;
  const int totalTiles    = B_ * tilesPerBatch;
  const int waveId        = blockIdx.x * 4 + w;
  const int numWaves      = gridDim.x * 4;

  for (int t = waveId; t < totalTiles; t += numWaves) {
    const int b  = t / tilesPerBatch;
    const int e0 = (t - b * tilesPerBatch) * 16;
    const float* nodeB = node  + (size_t)b * N_ * ND;
    const float* edgeB = eattr + (size_t)b * E_ * ED;
    float*       aggB  = agg   + (size_t)b * N_ * HD;

    if (lane < 16) sR[w][lane] = eidx[e0 + lane];
    __builtin_amdgcn_wave_barrier();

    const int recv = sR[w][m];
    const int sent = eidx[E_ + e0 + m];

    // Build f16 input tile: cols [0,64)=node[recv], [64,128)=node[sent], [128,160)=edge
    _Float16* xr = &sX[w][m * F1];
    const float4* nS = (const float4*)(nodeB + (size_t)sent * ND);
    if (hs == 0) {
      const float4* nR = (const float4*)(nodeB + (size_t)recv * ND);
      for (int i = 0; i < 16; ++i) cvt4(xr + i * 4, nR[i]);
      for (int i = 0; i < 4;  ++i) cvt4(xr + 64 + i * 4, nS[i]);
    } else {
      for (int i = 0; i < 12; ++i) cvt4(xr + 80 + i * 4, nS[4 + i]);
      const float4* eA = (const float4*)(edgeB + (size_t)(e0 + m) * ED);
      for (int i = 0; i < 8;  ++i) cvt4(xr + 128 + i * 4, eA[i]);
    }
    __builtin_amdgcn_wave_barrier();

    // ---- layer 1: h = relu(X @ W1^T + b1) ----
    v16h aF[5];
    for (int kc = 0; kc < 5; ++kc) aF[kc] = ldsA(xr, kc * 32, hs);

    for (int nc = 0; nc < 4; ++nc) {
      v8f acc;
      for (int i = 0; i < 8; ++i) acc[i] = bias1[nc];
      const _Float16* wrow = &sW1[(nc * 16 + m) * F1];
      for (int kc = 0; kc < 5; ++kc) {
        v16h bF = ldsB(wrow, kc * 32, hs);
        acc = __builtin_amdgcn_wmma_f32_16x16x32_f16(
            false, aF[kc], false, bF, (short)0, acc, false, false);
      }
      // C-layout: VGPR r, lane half -> row M = r + 8*hs, col N = nc*16 + m
      for (int r = 0; r < 8; ++r) {
        float v = acc[r] > 0.f ? acc[r] : 0.f;
        sH[w][(r + hs * 8) * HD + nc * 16 + m] = (_Float16)v;
      }
    }
    __builtin_amdgcn_wave_barrier();

    // ---- layer 2: msg = h @ W2^T + b2, scatter-add ----
    const _Float16* hrow = &sH[w][m * HD];
    v16h a2[2];
    a2[0] = ldsA(hrow, 0, hs);
    a2[1] = ldsA(hrow, 32, hs);

    int rIdx[8];
    for (int r = 0; r < 8; ++r) rIdx[r] = sR[w][r + hs * 8];

    for (int nc = 0; nc < 4; ++nc) {
      v8f acc;
      for (int i = 0; i < 8; ++i) acc[i] = bias2[nc];
      const _Float16* wrow = &sW2[(nc * 16 + m) * HD];
      for (int kc = 0; kc < 2; ++kc) {
        v16h bF = ldsB(wrow, kc * 32, hs);
        acc = __builtin_amdgcn_wmma_f32_16x16x32_f16(
            false, a2[kc], false, bF, (short)0, acc, false, false);
      }
      const int col = nc * 16 + m;
      for (int r = 0; r < 8; ++r) {
        atomicAdd(aggB + (size_t)rIdx[r] * HD + col, acc[r]);  // L2-resident f32 atomics
      }
    }
  }
}

// ---------------------------------------------------------------------------
// Node kernel: out = relu([node | agg] @ Wu^T + bu), 16-node tiles per wave
// ---------------------------------------------------------------------------
__launch_bounds__(128)
__global__ void gnn_node_kernel(const float* __restrict__ node,
                                const float* __restrict__ agg,
                                const float* __restrict__ Wu,
                                const float* __restrict__ bu,
                                float* __restrict__ out)
{
  __shared__ __align__(16) _Float16 sW[HD * F2];      // 16 KB
  __shared__ __align__(16) _Float16 sU[4][16 * F2];   // 16 KB

  const int tid = threadIdx.x;
  for (int i = tid; i < HD * F2; i += 128) sW[i] = (_Float16)Wu[i];
  __syncthreads();

  const int lane = tid & 31;
  const int w    = tid >> 5;
  const int m    = lane & 15;
  const int hs   = lane >> 4;

  float bias[4];
  for (int nc = 0; nc < 4; ++nc) bias[nc] = bu[nc * 16 + m];

  const int tilesPerBatch = N_ / 16;
  const int totalTiles    = B_ * tilesPerBatch;
  const int waveId        = blockIdx.x * 4 + w;
  const int numWaves      = gridDim.x * 4;

  for (int t = waveId; t < totalTiles; t += numWaves) {
    const int b  = t / tilesPerBatch;
    const int n0 = (t - b * tilesPerBatch) * 16;
    const float* nodeB = node + (size_t)b * N_ * ND;
    const float* aggB  = agg  + (size_t)b * N_ * HD;
    float*       outB  = out  + (size_t)b * N_ * HD;

    _Float16* ur = &sU[w][m * F2];
    if (hs == 0) {
      const float4* p = (const float4*)(nodeB + (size_t)(n0 + m) * ND);
      for (int i = 0; i < 16; ++i) cvt4(ur + i * 4, p[i]);
    } else {
      const float4* p = (const float4*)(aggB + (size_t)(n0 + m) * HD);
      for (int i = 0; i < 16; ++i) cvt4(ur + 64 + i * 4, p[i]);
    }
    __builtin_amdgcn_wave_barrier();

    v16h aF[4];
    for (int kc = 0; kc < 4; ++kc) aF[kc] = ldsA(ur, kc * 32, hs);

    for (int nc = 0; nc < 4; ++nc) {
      v8f acc;
      for (int i = 0; i < 8; ++i) acc[i] = bias[nc];
      const _Float16* wrow = &sW[(nc * 16 + m) * F2];
      for (int kc = 0; kc < 4; ++kc) {
        v16h bF = ldsB(wrow, kc * 32, hs);
        acc = __builtin_amdgcn_wmma_f32_16x16x32_f16(
            false, aF[kc], false, bF, (short)0, acc, false, false);
      }
      const int col = nc * 16 + m;
      for (int r = 0; r < 8; ++r) {
        float v = acc[r] > 0.f ? acc[r] : 0.f;
        outB[(size_t)(n0 + r + hs * 8) * HD + col] = v;
      }
    }
  }
}

// ---------------------------------------------------------------------------
extern "C" void kernel_launch(void* const* d_in, const int* in_sizes, int n_in,
                              void* d_out, int out_size, void* d_ws, size_t ws_size,
                              hipStream_t stream) {
  const float* node  = (const float*)d_in[0];
  const int*   eidx  = (const int*)d_in[1];
  const float* eattr = (const float*)d_in[2];
  const float* W1    = (const float*)d_in[3];
  const float* b1    = (const float*)d_in[4];
  const float* W2    = (const float*)d_in[5];
  const float* b2    = (const float*)d_in[6];
  const float* Wu    = (const float*)d_in[7];
  const float* bu    = (const float*)d_in[8];
  float* out = (float*)d_out;
  float* agg = (float*)d_ws;   // B*N*HD f32 = 20.5 MB scratch (L2-resident)

  const size_t aggBytes = (size_t)B_ * N_ * HD * sizeof(float);
  hipMemsetAsync(agg, 0, aggBytes, stream);

  // 160000 edge tiles over 5120 waves; 5000 node tiles over 1280 waves
  gnn_edge_kernel<<<1280, 128, 0, stream>>>(node, eidx, eattr, W1, b1, W2, b2, agg);
  gnn_node_kernel<<<320, 128, 0, stream>>>(node, agg, Wu, bu, out);
}
